// ChannelMambaFusionBlock_Mask_807453852221
// MI455X (gfx1250) — matrix operand order repaired and checked by reading, not compiled, not run on hardware
//
#include <hip/hip_runtime.h>
#include <hip/hip_bf16.h>
#include <math.h>
#include <stdint.h>

// ---------------- problem constants ----------------
#define BB 2
#define CC 96
#define HH 128
#define WW 128
#define LL (HH*WW)          // 16384
#define EE 192
#define GG 24
#define HEADS 16
#define CPH 12              // c = E/HEAD
#define NST 12              // D_STATE
#define HD 384              // MLP hidden

typedef __attribute__((ext_vector_type(16))) __bf16 v16bf;
typedef __attribute__((ext_vector_type(8)))  float  v8f;
typedef __attribute__((ext_vector_type(4)))  unsigned int v4u;
typedef __attribute__((ext_vector_type(8)))  int v8i_t;
typedef __attribute__((ext_vector_type(4)))  int v4i_t;

union Frag16 { v16bf v; uint4 q[2]; };
union FragC  { v8f  v; float f[8]; };

__device__ __forceinline__ unsigned short f2bf(float x) {
    unsigned u = __float_as_uint(x);
    unsigned r = ((u >> 16) & 1u) + 0x7FFFu;       // round-to-nearest-even
    return (unsigned short)((u + r) >> 16);
}

// ---------------- kernel 0: f32 -> bf16 weight convert ----------------
__global__ void k_f2bf(const float* __restrict__ src, unsigned short* __restrict__ dst, int n) {
    int i = blockIdx.x * blockDim.x + threadIdx.x;
    if (i < n) dst[i] = f2bf(src[i]);
}

// ---------------- kernel 1: channel-LN + grouped 1x1 inproj ----------------
// one thread per token (b,l); x channel-major (B,C,L); xr token-major (b,l,c);
// u_pre channel-major (B,E,L)
__global__ __launch_bounds__(256) void k_chanln_inproj(
    const float* __restrict__ x, const float* __restrict__ w, const float* __restrict__ bta,
    const float* __restrict__ ipw, float* __restrict__ xr, float* __restrict__ upre)
{
    int t = blockIdx.x * blockDim.x + threadIdx.x;
    if (t >= BB * LL) return;
    int b = t / LL, l = t % LL;
    const float* xb = x + (size_t)b * CC * LL + l;
    float s = 0.f, s2 = 0.f;
    for (int cc = 0; cc < CC; ++cc) { float v = xb[(size_t)cc * LL]; s += v; s2 += v * v; }
    float mu = s * (1.f / CC);
    float var = s2 * (1.f / CC) - mu * mu;
    float rs = rsqrtf(var + 1e-6f);
    float* xro = xr + (size_t)t * CC;
    float* up = upre + (size_t)b * EE * LL + l;
    for (int g = 0; g < GG; ++g) {
        float xn[4];
        #pragma unroll
        for (int j = 0; j < 4; ++j) {
            int c0 = g * 4 + j;
            float v = xb[(size_t)c0 * LL];
            v = (v - mu) * rs * w[c0] + bta[c0];
            xn[j] = v; xro[c0] = v;
        }
        #pragma unroll
        for (int i = 0; i < 8; ++i) {
            const float* wp = ipw + g * 32 + i * 4;
            float a = xn[0]*wp[0] + xn[1]*wp[1] + xn[2]*wp[2] + xn[3]*wp[3];
            up[(size_t)(g * 8 + i) * LL] = a;
        }
    }
}

// ---------------- kernel 2: depthwise 3x3 SAME + bias + SiLU ----------------
__global__ __launch_bounds__(256) void k_dwconv_silu(
    const float* __restrict__ in, const float* __restrict__ wt,
    const float* __restrict__ bs, float* __restrict__ out)
{
    int idx = blockIdx.x * blockDim.x + threadIdx.x;
    if (idx >= BB * EE * LL) return;
    int l = idx % LL;
    int e = (idx / LL) % EE;
    int b = idx / (EE * LL);
    int yy = l / WW, xx = l % WW;
    const float* ip = in + ((size_t)b * EE + e) * LL;
    const float* wp = wt + e * 9;
    float acc = bs[e];
    #pragma unroll
    for (int ky = 0; ky < 3; ++ky) {
        int y2 = yy + ky - 1;
        if ((unsigned)y2 < (unsigned)HH) {
            #pragma unroll
            for (int kx = 0; kx < 3; ++kx) {
                int x2 = xx + kx - 1;
                if ((unsigned)x2 < (unsigned)WW)
                    acc += ip[y2 * WW + x2] * wp[ky * 3 + kx];
            }
        }
    }
    out[idx] = acc * (1.f / (1.f + __expf(-acc)));
}

// ---------------- kernel 3: SSM (x_dbl, dt, 12-step scan, head-LN, outproj+res) ----------------
// one wave32 per token; lane = (stream k, head d). 8 tokens per 256-thread block.
__global__ __launch_bounds__(256) void k_ssm(
    const float* __restrict__ ur, const float* __restrict__ ue,
    const float* __restrict__ xprojw, const float* __restrict__ dtw_, const float* __restrict__ dtb_,
    const float* __restrict__ alogs, const float* __restrict__ ds_,
    const float* __restrict__ on1w, const float* __restrict__ on1b,
    const float* __restrict__ on2w, const float* __restrict__ on2b,
    const float* __restrict__ outw_r, const float* __restrict__ outw_e,
    const float* __restrict__ xr_all, float* __restrict__ ry_all)
{
    __shared__ float xpL[800];          // xproj_w (2,25,16)
    __shared__ float owL[1536];         // outproj (2,24,4,8)
    __shared__ float xsL[8][384];       // xs per token: [k][d*12+ci]
    __shared__ float xdL[8][600];       // x_dbl per token: [k][r*12+ci]
    __shared__ float yL[8][384];        // LN'd y per token: [k][e]

    int tid = threadIdx.x;
    for (int i = tid; i < 800; i += 256) xpL[i] = xprojw[i];
    for (int i = tid; i < 768; i += 256) { owL[i] = outw_r[i]; owL[768 + i] = outw_e[i]; }

    int wid = tid >> 5, lane = tid & 31;
    int tk = blockIdx.x * 8 + wid;       // token id, grid exact
    int b = tk / LL, l = tk % LL;
    int k = lane >> 4, d = lane & 15;

    const float* ub = (k == 0 ? ur : ue) + ((size_t)b * EE) * LL + l;
    float u[12];
    #pragma unroll
    for (int ci = 0; ci < 12; ++ci) {
        u[ci] = ub[(size_t)(d * 12 + ci) * LL];
        xsL[wid][k * 192 + d * 12 + ci] = u[ci];
    }
    __syncthreads();

    // x_dbl[k][r][ci] = sum_d xs[k][d][ci] * xproj_w[k][r][d]   (600 outputs / 32 lanes)
    for (int o = lane; o < 600; o += 32) {
        int kk2 = o / 300; int rem = o % 300; int r = rem / 12; int ci = rem % 12;
        const float* xp = xpL + kk2 * 400 + r * 16;
        const float* xsp = &xsL[wid][kk2 * 192 + ci];
        float a = 0.f;
        #pragma unroll
        for (int dd = 0; dd < 16; ++dd) a += xsp[dd * 12] * xp[dd];
        xdL[wid][o] = a;
    }
    __syncthreads();

    float dtw = dtw_[k * 16 + d], dtb = dtb_[k * 16 + d], Dv = ds_[k * 16 + d];
    float A[12];
    #pragma unroll
    for (int n = 0; n < 12; ++n) A[n] = -__expf(alogs[(k * 16 + d) * 12 + n]);
    float h[12];
    #pragma unroll
    for (int n = 0; n < 12; ++n) h[n] = 0.f;
    float yv[12];
    const float* xdk = &xdL[wid][k * 300];
    const float* xdo = &xdL[wid][(1 - k) * 300];
    #pragma unroll
    for (int ci = 0; ci < 12; ++ci) {
        float xln = dtw * xdk[ci] + dtb;                      // r=0 row is dts
        float dtv = (xln > 20.f) ? xln : log1pf(__expf(xln)); // softplus
        float du = dtv * u[ci];
        float y = 0.f;
        #pragma unroll
        for (int n = 0; n < 12; ++n) {
            float Bn = xdk[(1 + n) * 12 + ci];
            float Cn = xdo[(13 + n) * 12 + ci];              // cross-stream C
            h[n] = __expf(dtv * A[n]) * h[n] + du * Bn;
            y += h[n] * Cn;
        }
        yv[ci] = y + Dv * u[ci];
    }
    // LN over ci (eps 1e-5)
    float s = 0.f, s2 = 0.f;
    #pragma unroll
    for (int ci = 0; ci < 12; ++ci) { s += yv[ci]; s2 += yv[ci] * yv[ci]; }
    float mu = s * (1.f / 12.f);
    float rs = rsqrtf(s2 * (1.f / 12.f) - mu * mu + 1e-5f);
    const float* onw = (k == 0) ? on1w : on2w;
    const float* onb = (k == 0) ? on1b : on2b;
    #pragma unroll
    for (int ci = 0; ci < 12; ++ci)
        yL[wid][k * 192 + d * 12 + ci] = (yv[ci] - mu) * rs * onw[ci] + onb[ci];
    __syncthreads();

    // outproj (gconv1x1 E->C, groups of 8->4) + residual xr; ry token-major [s][b][l][c]
    #pragma unroll
    for (int j = 0; j < 6; ++j) {
        int o = lane + 32 * j;            // 0..191 = 2 streams * 96 couts
        int k2 = o / 96, cout = o % 96;
        int g = cout >> 2, co = cout & 3;
        const float* wp = owL + k2 * 768 + g * 32 + co * 8;
        const float* yp = &yL[wid][k2 * 192 + g * 8];
        float a = 0.f;
        #pragma unroll
        for (int jj = 0; jj < 8; ++jj) a += yp[jj] * wp[jj];
        size_t tok = (size_t)k2 * BB * LL + (size_t)b * LL + l;
        a += xr_all[tok * CC + cout];
        ry_all[tok * CC + cout] = a;
    }
}

// ---------------- kernel 4: channel-LN -> bf16 (pre-MLP) ----------------
// one wave per (stream, token); lane holds 3 channels; shfl_xor reduction
__global__ __launch_bounds__(256) void k_ln_bf16(
    const float* __restrict__ ry_all,
    const float* __restrict__ n1w, const float* __restrict__ n1b,
    const float* __restrict__ n2w, const float* __restrict__ n2b,
    unsigned short* __restrict__ zbf_all)
{
    int wid = threadIdx.x >> 5, lane = threadIdx.x & 31;
    int wv = blockIdx.x * 8 + wid;       // 0 .. 2*B*L-1
    int s = wv >> 15;                    // B*L = 32768
    int tk = wv & 32767;
    const float* rp = ry_all + ((size_t)s * BB * LL + tk) * CC;
    float v[3]; float sm = 0.f, sq = 0.f;
    #pragma unroll
    for (int j = 0; j < 3; ++j) { v[j] = rp[lane * 3 + j]; sm += v[j]; sq += v[j] * v[j]; }
    #pragma unroll
    for (int m = 16; m >= 1; m >>= 1) {
        sm += __shfl_xor(sm, m, 32);
        sq += __shfl_xor(sq, m, 32);
    }
    float mu = sm * (1.f / CC);
    float rs = rsqrtf(sq * (1.f / CC) - mu * mu + 1e-6f);
    const float* nw = s ? n2w : n1w;
    const float* nb = s ? n2b : n1b;
    unsigned short* zp = zbf_all + ((size_t)s * BB * LL + tk) * CC;
    #pragma unroll
    for (int j = 0; j < 3; ++j) {
        int c = lane * 3 + j;
        zp[c] = f2bf((v[j] - mu) * rs * nw[c] + nb[c]);
    }
}

// ---------------- kernel 5: MLP via WMMA bf16 (fc1 -> GELU -> fc2 -> +bias +res) ----------------
// 16-token tile per WG, 6 waves (192 threads). fc1: wave owns 4 N-tiles (24 total, N=384).
// fc2: wave owns 1 N-tile (6 total, N=96). Output written channel-major into d_out.
// A tile (16 tokens x 96 bf16, 3072B contiguous) staged into LDS via the Tensor Data Mover
// when available; manual cooperative copy as fallback.
__global__ __launch_bounds__(192) void k_mlp(
    const unsigned short* __restrict__ zbf_all, const float* __restrict__ ry_all,
    const unsigned short* __restrict__ w1bf_all, const float* __restrict__ fc1b_r, const float* __restrict__ fc1b_e,
    const unsigned short* __restrict__ w2bf_all, const float* __restrict__ fc2b_r, const float* __restrict__ fc2b_e,
    float* __restrict__ out)
{
    int s = blockIdx.y;
    int t0 = blockIdx.x * 16;            // token base within stream (b*L+l flat)
    __shared__ __align__(16) unsigned short Azb[16 * CC];   // 16 tokens x 96 (bf16)
    __shared__ __align__(16) unsigned short Hh[16 * HD];    // 16 tokens x 384 (bf16)

    int tid = threadIdx.x, wid = tid >> 5, lane = tid & 31;
    const unsigned short* srcA16 = zbf_all + ((size_t)s * BB * LL + t0) * CC;

#if __has_builtin(__builtin_amdgcn_tensor_load_to_lds) && __has_builtin(__builtin_amdgcn_s_wait_tensorcnt)
    // ---- TDM path: one 1D tensor load (1536 x 2B) into LDS, issued by one wave ----
    if (tid == 0) {
        unsigned ldsOff = (unsigned)(uintptr_t)(void*)Azb;           // LDS aperture: low 32 bits
        unsigned long long ga = (unsigned long long)(uintptr_t)srcA16;
        v4u g0;
        g0[0] = 1u;                                   // count=1, user descriptor, no gather
        g0[1] = ldsOff;                               // lds_addr (bytes)
        g0[2] = (unsigned)(ga & 0xFFFFFFFFull);       // global_addr[31:0]
        g0[3] = (unsigned)((ga >> 32) & 0x1FFFFFFull) // global_addr[56:32]
              | (2u << 30);                           // type = 2 ("image")
        v8i_t g1;
        g1[0] = 0x00010000;       // wg_mask=0, data_size=1 (2 bytes)
        g1[1] = (int)(1536u << 16); // tensor_dim0[15:0] in bits[63:48]
        g1[2] = 0x00010000;       // tensor_dim0 hi=0, tensor_dim1=1
        g1[3] = (int)(1536u << 16); // tensor_dim1 hi=0, tile_dim0=1536
        g1[4] = 1;                // tile_dim1=1, tile_dim2=0
        g1[5] = 1536;             // tensor_dim0_stride lo
        g1[6] = 0;                // stride0 hi, stride1 lo
        g1[7] = 0;                // stride1 hi
        v4i_t z4 = {0, 0, 0, 0};
#if defined(__clang_major__) && (__clang_major__ >= 23)
        v8i_t z8 = {0, 0, 0, 0, 0, 0, 0, 0};
        __builtin_amdgcn_tensor_load_to_lds(g0, g1, z4, z4, z8, 0);
#else
        __builtin_amdgcn_tensor_load_to_lds(g0, g1, z4, z4, 0);
#endif
        __builtin_amdgcn_s_wait_tensorcnt(0);
    }
#else
    // ---- fallback: cooperative dword copy (1536 halves = 768 dwords) ----
    {
        const unsigned* srcA = (const unsigned*)srcA16;
        unsigned* dstA = (unsigned*)Azb;
        for (int i = tid; i < 768; i += 192) dstA[i] = srcA[i];
    }
#endif
    __syncthreads();

    const unsigned short* w1 = w1bf_all + (size_t)s * (HD * CC);
    const float* b1 = s ? fc1b_e : fc1b_r;
    int m_lo = lane & 15, hi = lane >> 4;
    int kbA = hi * 8;      // A layout: K = kbA + h (+8 for h>=8)
    int kbB = hi * 8;      // B layout: K = kbB + h (+8 for h>=8)

    // prefetch this wave's weight rows (global_prefetch path)
    __builtin_prefetch(w1 + (size_t)((wid * 4) * 16 + m_lo) * CC, 0, 1);
    __builtin_prefetch(w2bf_all + (size_t)s * (CC * HD) + (size_t)(wid * 16 + m_lo) * HD, 0, 1);

    FragC acc[4];
    #pragma unroll
    for (int t = 0; t < 4; ++t)
        #pragma unroll
        for (int r = 0; r < 8; ++r) acc[t].f[r] = 0.f;

    for (int kk = 0; kk < CC; kk += 32) {
        Frag16 af;
        const uint4* ap = (const uint4*)(Azb + m_lo * CC + kk + kbA);
        af.q[0] = ap[0]; af.q[1] = ap[2];     // halves {kbA..+7} and {kbA+16..+23}
        #pragma unroll
        for (int t = 0; t < 4; ++t) {
            int n = (wid * 4 + t) * 16 + m_lo;
            const uint4* bp = (const uint4*)(w1 + (size_t)n * CC + kk + kbB);
            Frag16 bf; bf.q[0] = bp[0]; bf.q[1] = bp[2];   // halves {kbB..+7} and {kbB+16..+23}
            acc[t].v = __builtin_amdgcn_wmma_f32_16x16x32_bf16(
                false, af.v, false, bf.v, (short)0, acc[t].v, false, false);
        }
    }
    // bias + exact GELU + store hidden (bf16) via D layout (m = r + 8*hi, n = tile*16 + m_lo)
    #pragma unroll
    for (int t = 0; t < 4; ++t) {
        int n = (wid * 4 + t) * 16 + m_lo;
        float bias = b1[n];
        #pragma unroll
        for (int r = 0; r < 8; ++r) {
            int m = r + 8 * hi;
            float v = acc[t].f[r] + bias;
            v = 0.5f * v * (1.f + erff(v * 0.70710678118f));
            Hh[m * HD + n] = f2bf(v);
        }
    }
    __syncthreads();

    // fc2: N=96, wave owns tile wid
    const unsigned short* w2 = w2bf_all + (size_t)s * (CC * HD);
    const float* b2 = s ? fc2b_e : fc2b_r;
    FragC c2;
    #pragma unroll
    for (int r = 0; r < 8; ++r) c2.f[r] = 0.f;
    int n2i = wid * 16 + m_lo;
    for (int kk = 0; kk < HD; kk += 32) {
        Frag16 af;
        const uint4* ap = (const uint4*)(Hh + m_lo * HD + kk + kbA);
        af.q[0] = ap[0]; af.q[1] = ap[2];
        const uint4* bp = (const uint4*)(w2 + (size_t)n2i * HD + kk + kbB);
        Frag16 bf; bf.q[0] = bp[0]; bf.q[1] = bp[2];
        c2.v = __builtin_amdgcn_wmma_f32_16x16x32_bf16(
            false, af.v, false, bf.v, (short)0, c2.v, false, false);
    }
    float bias2 = b2[n2i];
    const float* ryp = ry_all + ((size_t)s * BB * LL + t0) * CC;
    float* op = out + (size_t)s * BB * CC * LL;
    int bb = t0 / LL, l0 = t0 % LL;      // all 16 tokens share b (L % 16 == 0)
    #pragma unroll
    for (int r = 0; r < 8; ++r) {
        int m = r + 8 * hi;
        float v = c2.f[r] + bias2 + ryp[m * CC + n2i];
        op[((size_t)bb * CC + n2i) * LL + (l0 + m)] = v;
    }
}

// ---------------- host launch ----------------
extern "C" void kernel_launch(void* const* d_in, const int* in_sizes, int n_in,
                              void* d_out, int out_size, void* d_ws, size_t ws_size,
                              hipStream_t stream) {
    (void)in_sizes; (void)n_in; (void)out_size; (void)ws_size;
    const float* x_rgb   = (const float*)d_in[0];
    const float* x_e     = (const float*)d_in[1];
    const float* in1w    = (const float*)d_in[2];
    const float* in1b    = (const float*)d_in[3];
    const float* in2w    = (const float*)d_in[4];
    const float* in2b    = (const float*)d_in[5];
    const float* ipw_r   = (const float*)d_in[6];
    const float* ipw_e   = (const float*)d_in[7];
    const float* cw_r    = (const float*)d_in[8];
    const float* cb_r    = (const float*)d_in[9];
    const float* cw_e    = (const float*)d_in[10];
    const float* cb_e    = (const float*)d_in[11];
    const float* xprojw  = (const float*)d_in[12];
    const float* dtw     = (const float*)d_in[13];
    const float* dtb     = (const float*)d_in[14];
    const float* alogs   = (const float*)d_in[15];
    const float* Ds      = (const float*)d_in[16];
    const float* on1w    = (const float*)d_in[17];
    const float* on1b    = (const float*)d_in[18];
    const float* on2w    = (const float*)d_in[19];
    const float* on2b    = (const float*)d_in[20];
    const float* opw_r   = (const float*)d_in[21];
    const float* opw_e   = (const float*)d_in[22];
    const float* n1w     = (const float*)d_in[23];
    const float* n1b     = (const float*)d_in[24];
    const float* n2w     = (const float*)d_in[25];
    const float* n2b     = (const float*)d_in[26];
    const float* fc1w_r  = (const float*)d_in[27];
    const float* fc1b_r  = (const float*)d_in[28];
    const float* fc2w_r  = (const float*)d_in[29];
    const float* fc2b_r  = (const float*)d_in[30];
    const float* fc1w_e  = (const float*)d_in[31];
    const float* fc1b_e  = (const float*)d_in[32];
    const float* fc2w_e  = (const float*)d_in[33];
    const float* fc2b_e  = (const float*)d_in[34];

    // workspace layout (floats). Phase reuse:
    //   [XR: 2*B*L*C][UPRE/RY: 2*B*E*L][UR/ZBF: 2*B*E*L][WBF: bf16 weights]
    const size_t SZ_XR   = (size_t)2 * BB * LL * CC;   //  6,291,456
    const size_t SZ_UPRE = (size_t)2 * BB * EE * LL;   // 12,582,912
    float* ws = (float*)d_ws;
    float* xr_all  = ws;                               // [s][b][l][c]
    float* upre    = ws + SZ_XR;                       // [s][b][e][l]
    float* ur      = ws + SZ_XR + SZ_UPRE;             // [s][b][e][l]
    float* ry_all  = upre;                             // reuse: [s][b][l][c]
    unsigned short* zbf_all = (unsigned short*)ur;     // reuse: [s][b][l][c] bf16
    unsigned short* wbf = (unsigned short*)(ws + SZ_XR + 2 * SZ_UPRE);
    unsigned short* w1bf = wbf;                        // [s][384*96]
    unsigned short* w2bf = wbf + 2 * (HD * CC);        // [s][96*384]

    const int NW = HD * CC;  // 36864
    k_f2bf<<<(NW + 255) / 256, 256, 0, stream>>>(fc1w_r, w1bf,          NW);
    k_f2bf<<<(NW + 255) / 256, 256, 0, stream>>>(fc1w_e, w1bf + NW,     NW);
    k_f2bf<<<(NW + 255) / 256, 256, 0, stream>>>(fc2w_r, w2bf,          NW);
    k_f2bf<<<(NW + 255) / 256, 256, 0, stream>>>(fc2w_e, w2bf + NW,     NW);

    const int BL = BB * LL;  // 32768
    k_chanln_inproj<<<BL / 256, 256, 0, stream>>>(x_rgb, in1w, in1b, ipw_r,
                                                  xr_all,                 upre);
    k_chanln_inproj<<<BL / 256, 256, 0, stream>>>(x_e,   in2w, in2b, ipw_e,
                                                  xr_all + (size_t)BL * CC, upre + (size_t)BB * EE * LL);

    const int BEL = BB * EE * LL;  // 6,291,456
    k_dwconv_silu<<<BEL / 256, 256, 0, stream>>>(upre,                        cw_r, cb_r, ur);
    k_dwconv_silu<<<BEL / 256, 256, 0, stream>>>(upre + (size_t)BB * EE * LL, cw_e, cb_e, ur + (size_t)BB * EE * LL);

    k_ssm<<<BL / 8, 256, 0, stream>>>(ur, ur + (size_t)BB * EE * LL,
                                      xprojw, dtw, dtb, alogs, Ds,
                                      on1w, on1b, on2w, on2b,
                                      opw_r, opw_e, xr_all, ry_all);

    k_ln_bf16<<<(2 * BL) / 8, 256, 0, stream>>>(ry_all, n1w, n1b, n2w, n2b, zbf_all);

    dim3 gmlp(BL / 16, 2);
    k_mlp<<<gmlp, 192, 0, stream>>>(zbf_all, ry_all,
                                    w1bf, fc1b_r, fc1b_e,
                                    w2bf, fc2b_r, fc2b_e,
                                    (float*)d_out);
}
